// FusedQKNormAttention_90383291777468
// MI455X (gfx1250) — matrix operand order repaired and checked by reading, not compiled, not run on hardware
//
#include <hip/hip_runtime.h>
#include <hip/hip_bf16.h>

typedef __attribute__((ext_vector_type(16))) _Float16 v16h;
typedef __attribute__((ext_vector_type(8)))  _Float16 v8h;
typedef __attribute__((ext_vector_type(8)))  float    v8f;

// ---------------- problem constants ----------------
constexpr int BATCH   = 2;
constexpr int SEQ     = 2048;
constexpr int HIDDEN  = 2048;
constexpr int NHEADS  = 32;
constexpr int KVHEADS = 8;
constexpr int HEADDIM = 64;
constexpr int MROWS   = BATCH * SEQ;  // 4096 flattened rows

// workspace layout (in _Float16 elements)
constexpr size_t OFF_XH = 0;                                           // [4096,2048]
constexpr size_t OFF_WQ = OFF_XH + (size_t)MROWS * HIDDEN;             // [2048,2048]
constexpr size_t OFF_WK = OFF_WQ + (size_t)HIDDEN * HIDDEN;            // [512,2048]
constexpr size_t OFF_WV = OFF_WK + (size_t)(KVHEADS*HEADDIM) * HIDDEN;
constexpr size_t OFF_WO = OFF_WV + (size_t)(KVHEADS*HEADDIM) * HIDDEN;
constexpr size_t OFF_QN = OFF_WO + (size_t)HIDDEN * HIDDEN;            // [B,32,S,64]
constexpr size_t OFF_KN = OFF_QN + (size_t)BATCH*NHEADS*SEQ*HEADDIM;   // [B,8,S,64]
constexpr size_t OFF_VT = OFF_KN + (size_t)BATCH*KVHEADS*SEQ*HEADDIM;  // [B,8,64,S] (transposed!)
constexpr size_t OFF_AO = OFF_VT + (size_t)BATCH*KVHEADS*SEQ*HEADDIM;  // [B*S,2048]

// ---------------- WMMA helpers ----------------
__device__ inline v8f wmma16(v16h a, v16h b, v8f c) {
  return __builtin_amdgcn_wmma_f32_16x16x32_f16(
      false, a, false, b, (short)0, c, false, false);
}

// A fragment (16x32 f16), row-major source, ISA 16-bit A layout:
// lane m = lane&15 ; VGPR v pair base k = (v&3)*2 + (v>>2)*16 + (lane>>4)*8
// -> per lane: halves p[0..7] then p[16..23] (two 16B vector loads)
__device__ inline v16h load_a_rm(const _Float16* __restrict__ base, int ld,
                                 int m0, int k0, unsigned lane) {
  const unsigned half = lane >> 4, mr = lane & 15;
  const _Float16* p = base + (size_t)(m0 + mr) * ld + k0 + (half << 3);
  const v8h lo = *(const v8h*)(p);
  const v8h hi = *(const v8h*)(p + 16);
  v16h a;
#pragma unroll
  for (int i = 0; i < 8; ++i) { a[i] = lo[i]; a[8 + i] = hi[i]; }
  return a;
}

// B fragment (32x16) where B[k,n] = src[(n0+n)*ld + k0+k] (k contiguous).
// lane n = lane&15 ; same k order as A: halves p[0..7] then p[16..23], base k = (lane>>4)*8
// -> per lane: two 16B vector loads
__device__ inline v16h load_b_wt(const _Float16* __restrict__ base, int ld,
                                 int n0, int k0, unsigned lane) {
  const unsigned half = lane >> 4, nc = lane & 15;
  const _Float16* p = base + (size_t)(n0 + nc) * ld + k0 + (half << 3);
  const v8h lo = *(const v8h*)(p);
  const v8h hi = *(const v8h*)(p + 16);
  v16h b;
#pragma unroll
  for (int i = 0; i < 8; ++i) { b[i] = lo[i]; b[8 + i] = hi[i]; }
  return b;
}

__device__ inline float red16_max(float v) {
  v = fmaxf(v, __shfl_xor(v, 1, 32));
  v = fmaxf(v, __shfl_xor(v, 2, 32));
  v = fmaxf(v, __shfl_xor(v, 4, 32));
  v = fmaxf(v, __shfl_xor(v, 8, 32));
  return v;
}
__device__ inline float red16_sum(float v) {
  v += __shfl_xor(v, 1, 32);
  v += __shfl_xor(v, 2, 32);
  v += __shfl_xor(v, 4, 32);
  v += __shfl_xor(v, 8, 32);
  return v;
}

// ---------------- f32 -> f16 convert ----------------
__global__ void cvt_f32_f16(const float* __restrict__ src,
                            _Float16* __restrict__ dst, int n) {
  for (int i = blockIdx.x * blockDim.x + threadIdx.x; i < n;
       i += gridDim.x * blockDim.x)
    dst[i] = (_Float16)src[i];
}

// ---------------- wave-tile GEMM: Y = A(f16,[M,K]) * W(f16,[N,K])^T ----------
// Each wave computes a 16x64 tile (one head-width).
// MODE 0: store f32 row-major [M,N] to Of32
// MODE 1: fused per-64 RMSNorm then f16 store to heads layout [B,H,S,64]
// MODE 2: f16 store TRANSPOSED heads layout [B,H,64,S] (for V)
template <int MODE>
__global__ __launch_bounds__(256) void gemm_wmma(
    const _Float16* __restrict__ A, const _Float16* __restrict__ W,
    float* __restrict__ Of32, _Float16* __restrict__ Oh,
    const float* __restrict__ normw, int M, int N, int K, int S) {
  const int wave  = blockIdx.x * (blockDim.x >> 5) + (threadIdx.x >> 5);
  const unsigned lane = threadIdx.x & 31;
  const int colTiles = N >> 6;
  const int rowTiles = M >> 4;
  if (wave >= rowTiles * colTiles) return;
  const int tc = wave % colTiles, tr = wave / colTiles;
  const int m0 = tr << 4, n0 = tc << 6;

  v8f acc[4] = {};
  for (int k0 = 0; k0 < K; k0 += 32) {
    const v16h a  = load_a_rm(A, K, m0, k0, lane);
    const v16h b0 = load_b_wt(W, K, n0 +  0, k0, lane);
    const v16h b1 = load_b_wt(W, K, n0 + 16, k0, lane);
    const v16h b2 = load_b_wt(W, K, n0 + 32, k0, lane);
    const v16h b3 = load_b_wt(W, K, n0 + 48, k0, lane);
    acc[0] = wmma16(a, b0, acc[0]);
    acc[1] = wmma16(a, b1, acc[1]);
    acc[2] = wmma16(a, b2, acc[2]);
    acc[3] = wmma16(a, b3, acc[3]);
  }

  const unsigned half = lane >> 4, nc = lane & 15;
  if constexpr (MODE == 0) {
#pragma unroll
    for (int f = 0; f < 4; ++f)
#pragma unroll
      for (int i = 0; i < 8; ++i) {
        const int m = m0 + i + (int)(half << 3);
        Of32[(size_t)m * N + n0 + f * 16 + nc] = acc[f][i];
      }
  } else if constexpr (MODE == 1) {
    const int H = N >> 6;
    float rms[8];
#pragma unroll
    for (int i = 0; i < 8; ++i) {
      float ss = acc[0][i] * acc[0][i] + acc[1][i] * acc[1][i] +
                 acc[2][i] * acc[2][i] + acc[3][i] * acc[3][i];
      ss = red16_sum(ss);
      rms[i] = rsqrtf(ss * (1.0f / 64.0f) + 1e-6f);
    }
#pragma unroll
    for (int f = 0; f < 4; ++f) {
      const float w = normw[f * 16 + nc];
#pragma unroll
      for (int i = 0; i < 8; ++i) {
        const int m = m0 + i + (int)(half << 3);
        const int b = m / S, s = m % S;
        const size_t idx =
            (((size_t)(b * H + tc)) * S + s) * HEADDIM + f * 16 + nc;
        Oh[idx] = (_Float16)(acc[f][i] * rms[i] * w);
      }
    }
  } else {
    // MODE 2: transposed store Vt[b, h, d, s]; per lane the 8 rows i give
    // consecutive s -> one contiguous 16B (v8h) vector store.
    const int H = N >> 6;
#pragma unroll
    for (int f = 0; f < 4; ++f) {
      const int d = f * 16 + (int)nc;
      const int mbase = m0 + (int)(half << 3);
      const int b = mbase / S, sbase = mbase % S;  // 16-row tiles never cross batch
      _Float16* dst =
          Oh + (((size_t)(b * H + tc)) * HEADDIM + d) * S + sbase;
      v8h vals;
#pragma unroll
      for (int i = 0; i < 8; ++i) vals[i] = (_Float16)acc[f][i];
      *(v8h*)dst = vals;
    }
  }
}

// ---------------- flash attention: one wave per (b, h, 16-row q-tile) -------
__global__ __launch_bounds__(128) void flash_attn(
    const _Float16* __restrict__ Qn, const _Float16* __restrict__ Kn,
    const _Float16* __restrict__ Vt, _Float16* __restrict__ AO, int S) {
  const int qTiles = S >> 4;
  const int wid = blockIdx.x * (blockDim.x >> 5) + (threadIdx.x >> 5);
  const int total = BATCH * NHEADS * qTiles;
  if (wid >= total) return;
  const unsigned lane = threadIdx.x & 31;
  const int wslot = (threadIdx.x >> 5) & 3;

  const int qt  = wid % qTiles;
  const int h   = (wid / qTiles) % NHEADS;
  const int b   = wid / (qTiles * NHEADS);
  const int kvh = h >> 2;  // GQA group of 4

  const _Float16* Qp = Qn + ((size_t)(b * NHEADS  + h  ) * S) * HEADDIM;
  const _Float16* Kp = Kn + ((size_t)(b * KVHEADS + kvh) * S) * HEADDIM;
  const _Float16* Vp = Vt + ((size_t)(b * KVHEADS + kvh) * HEADDIM) * S;
  const int q0 = qt << 4;

  // Q tile [16,64] as two A fragments, kept in registers.
  const v16h aq0 = load_a_rm(Qp, HEADDIM, q0,  0, lane);
  const v16h aq1 = load_a_rm(Qp, HEADDIM, q0, 32, lane);

  v8f o[4] = {};
  float rmax[8], rsum[8];
#pragma unroll
  for (int i = 0; i < 8; ++i) { rmax[i] = -1e30f; rsum[i] = 0.0f; }

  // per-wave LDS slice for the C-layout -> A-layout re-shape of P
  __shared__ _Float16 P[4][16][40];  // 80B row pitch keeps 16B alignment

  const unsigned half = lane >> 4, nc = lane & 15;
  const float scl = 0.125f;  // 1/sqrt(64)

  for (int k0 = 0; k0 < q0 + 16; k0 += 32) {
    // scores for 32 keys: two 16x16 C fragments, each K=64 -> 2 WMMAs
    v8f s0 = {}, s1 = {};
    s0 = wmma16(aq0, load_b_wt(Kp, HEADDIM, k0,       0, lane), s0);
    s0 = wmma16(aq1, load_b_wt(Kp, HEADDIM, k0,      32, lane), s0);
    s1 = wmma16(aq0, load_b_wt(Kp, HEADDIM, k0 + 16,  0, lane), s1);
    s1 = wmma16(aq1, load_b_wt(Kp, HEADDIM, k0 + 16, 32, lane), s1);

    // online softmax over rows (row m = i + half*8, key j = lane column)
#pragma unroll
    for (int i = 0; i < 8; ++i) {
      const int mg = q0 + i + (int)(half << 3);
      const int j0 = k0 + (int)nc, j1 = j0 + 16;
      const float v0 = (j0 <= mg) ? s0[i] * scl : -1e30f;
      const float v1 = (j1 <= mg) ? s1[i] * scl : -1e30f;
      const float mx = red16_max(fmaxf(v0, v1));
      const float nm = fmaxf(rmax[i], mx);
      const float fct = __expf(rmax[i] - nm);
      const float p0 = __expf(v0 - nm);
      const float p1 = __expf(v1 - nm);
      rsum[i] = rsum[i] * fct + red16_sum(p0 + p1);
      rmax[i] = nm;
      o[0][i] *= fct; o[1][i] *= fct; o[2][i] *= fct; o[3][i] *= fct;
      P[wslot][i + (half << 3)][nc]      = (_Float16)p0;
      P[wslot][i + (half << 3)][nc + 16] = (_Float16)p1;
    }

    // re-load P as a 16x32 A fragment from LDS (same-wave DS ops are in-order)
    v16h ap;
    {
      const _Float16* pp = &P[wslot][lane & 15][half << 3];
      const v8h lo = *(const v8h*)(pp);
      const v8h hi = *(const v8h*)(pp + 16);
#pragma unroll
      for (int i = 0; i < 8; ++i) { ap[i] = lo[i]; ap[8 + i] = hi[i]; }
    }

    // O += P[16,32] x V[32,64]; Vt is [64,S] so B loads are contiguous in key
#pragma unroll
    for (int f = 0; f < 4; ++f) {
      const v16h bv = load_b_wt(Vp, S, f * 16, k0, lane);
      o[f] = wmma16(ap, bv, o[f]);
    }
  }

  // epilogue: normalize and scatter to [B, S, H*64] f16
#pragma unroll
  for (int i = 0; i < 8; ++i) {
    const float inv = 1.0f / rsum[i];
    const int s = q0 + i + (int)(half << 3);
    const size_t rowbase = ((size_t)b * S + s) * (size_t)HIDDEN + h * HEADDIM;
    AO[rowbase +  0 + nc] = (_Float16)(o[0][i] * inv);
    AO[rowbase + 16 + nc] = (_Float16)(o[1][i] * inv);
    AO[rowbase + 32 + nc] = (_Float16)(o[2][i] * inv);
    AO[rowbase + 48 + nc] = (_Float16)(o[3][i] * inv);
  }
}

// ---------------- launch ----------------
extern "C" void kernel_launch(void* const* d_in, const int* in_sizes, int n_in,
                              void* d_out, int out_size, void* d_ws,
                              size_t ws_size, hipStream_t stream) {
  const float* x   = (const float*)d_in[0];
  const float* Wq  = (const float*)d_in[1];
  const float* Wk  = (const float*)d_in[2];
  const float* Wv  = (const float*)d_in[3];
  const float* Wo  = (const float*)d_in[4];
  const float* qnw = (const float*)d_in[5];
  const float* knw = (const float*)d_in[6];
  float* out = (float*)d_out;

  _Float16* ws = (_Float16*)d_ws;
  _Float16* xh  = ws + OFF_XH;
  _Float16* wqh = ws + OFF_WQ;
  _Float16* wkh = ws + OFF_WK;
  _Float16* wvh = ws + OFF_WV;
  _Float16* woh = ws + OFF_WO;
  _Float16* qn  = ws + OFF_QN;
  _Float16* kn  = ws + OFF_KN;
  _Float16* vt  = ws + OFF_VT;
  _Float16* ao  = ws + OFF_AO;

  auto cvt = [&](const float* s, _Float16* d, int n) {
    int blocks = (n + 255) / 256;
    if (blocks > 4096) blocks = 4096;
    cvt_f32_f16<<<blocks, 256, 0, stream>>>(s, d, n);
  };
  cvt(x,  xh,  MROWS * HIDDEN);
  cvt(Wq, wqh, HIDDEN * HIDDEN);
  cvt(Wk, wkh, KVHEADS * HEADDIM * HIDDEN);
  cvt(Wv, wvh, KVHEADS * HEADDIM * HIDDEN);
  cvt(Wo, woh, HIDDEN * HIDDEN);

  // Q projection + q RMSNorm -> [B,32,S,64]
  {
    int waves = (MROWS / 16) * (HIDDEN / 64);
    gemm_wmma<1><<<(waves + 7) / 8, 256, 0, stream>>>(
        xh, wqh, nullptr, qn, qnw, MROWS, HIDDEN, HIDDEN, SEQ);
  }
  // K projection + k RMSNorm -> [B,8,S,64]
  {
    int N = KVHEADS * HEADDIM;
    int waves = (MROWS / 16) * (N / 64);
    gemm_wmma<1><<<(waves + 7) / 8, 256, 0, stream>>>(
        xh, wkh, nullptr, kn, knw, MROWS, N, HIDDEN, SEQ);
  }
  // V projection -> transposed [B,8,64,S]
  {
    int N = KVHEADS * HEADDIM;
    int waves = (MROWS / 16) * (N / 64);
    gemm_wmma<2><<<(waves + 7) / 8, 256, 0, stream>>>(
        xh, wvh, nullptr, vt, nullptr, MROWS, N, HIDDEN, SEQ);
  }
  // flash attention -> ao [B*S, 2048] f16
  {
    int waves = BATCH * NHEADS * (SEQ / 16);
    flash_attn<<<(waves + 3) / 4, 128, 0, stream>>>(qn, kn, vt, ao, SEQ);
  }
  // output projection -> d_out f32
  {
    int waves = (MROWS / 16) * (HIDDEN / 64);
    gemm_wmma<0><<<(waves + 7) / 8, 256, 0, stream>>>(
        ao, woh, out, nullptr, nullptr, MROWS, HIDDEN, HIDDEN, SEQ);
  }
}
